// MLAttention_50646254355275
// MI455X (gfx1250) — matrix operand order repaired and checked by reading, not compiled, not run on hardware
//
#include <hip/hip_runtime.h>

// ---------------------------------------------------------------------------
// MLA forward (DeepSeek-style) for gfx1250: bf16 WMMA GEMMs + flash attention
// GEMM uses CDNA5 async global->LDS DMA with double-buffered tiles.
// ---------------------------------------------------------------------------

typedef unsigned short u16;
typedef __attribute__((ext_vector_type(16))) __bf16        v16bf;
typedef __attribute__((ext_vector_type(8)))  float         v8f;
typedef __attribute__((ext_vector_type(4)))  unsigned int  u32x4;

union Frag {            // one WMMA 16-bit A/B operand: 16 bf16 per lane
    u32x4 u[2];
    v16bf v;
};

__device__ inline v8f zero8() {
    v8f v = {0.f, 0.f, 0.f, 0.f, 0.f, 0.f, 0.f, 0.f};
    return v;
}

// fp32 -> bf16 bits, round-to-nearest-even
__device__ inline u16 f2bf(float f) {
    unsigned u = __float_as_uint(f);
    unsigned r = 0x7FFFu + ((u >> 16) & 1u);
    return (u16)((u + r) >> 16);
}

// CDNA5 async global->LDS copy (ASYNCcnt-tracked), 16B per lane.
// ldsOff: byte offset into wave's LDS (flat-addr low 32 bits).
__device__ inline void async_b128(unsigned ldsOff, const void* g) {
    asm volatile("global_load_async_to_lds_b128 %0, %1, off"
                 :: "v"(ldsOff), "v"(g) : "memory");
}
__device__ inline void wait_async0() {
    asm volatile("s_wait_asynccnt 0x0" ::: "memory");
}

// ---------------------------------------------------------------------------
// fp32 -> bf16 elementwise
// ---------------------------------------------------------------------------
__global__ void f32_to_bf16_kernel(const float* __restrict__ in,
                                   u16* __restrict__ out, size_t n) {
    size_t i = (size_t)blockIdx.x * blockDim.x + threadIdx.x;
    if (i < n) out[i] = f2bf(in[i]);
}

// ---------------------------------------------------------------------------
// fp32 [K x N] -> bf16 [N x K] tiled transpose (weights -> Bt layout)
// ---------------------------------------------------------------------------
__global__ __launch_bounds__(256) void transpose_f32_to_bf16(
        const float* __restrict__ in, u16* __restrict__ out, int K, int N) {
    __shared__ float tile[32][33];
    int k0 = blockIdx.y * 32, n0 = blockIdx.x * 32;
    int tx = threadIdx.x & 31, ty = threadIdx.x >> 5;   // ty 0..7
    #pragma unroll
    for (int i = ty; i < 32; i += 8) {
        int k = k0 + i, n = n0 + tx;
        tile[i][tx] = (k < K && n < N) ? in[(size_t)k * N + n] : 0.f;
    }
    __syncthreads();
    #pragma unroll
    for (int i = ty; i < 32; i += 8) {
        int n = n0 + i, k = k0 + tx;
        if (n < N && k < K) out[(size_t)n * K + k] = f2bf(tile[tx][i]);
    }
}

// ---------------------------------------------------------------------------
// C[M,N] fp32 = A[M,K] bf16 (row major) @ Bt[N,K] bf16 (row major, = B^T)
// BM=128, BN=64, BK=32. 256 threads = 8 waves, wave (wm 0..3, wn 0..1),
// each wave owns a 32x32 patch = 2x2 WMMA tiles.
// Tiles staged with global_load_async_to_lds_b128, double buffered.
// ---------------------------------------------------------------------------
#define GBM 128
#define GBN 64
#define GBK 32

__global__ __launch_bounds__(256) void gemm_bf16_f32(
        const u16* __restrict__ A, const u16* __restrict__ Bt,
        float* __restrict__ C, int M, int N, int K) {
    __shared__ u16 As[2][GBM][GBK + 8];
    __shared__ u16 Bs[2][GBN][GBK + 8];

    const int tid  = threadIdx.x;
    const int lane = tid & 31, wid = tid >> 5;
    const int wm = wid & 3, wn = wid >> 2;
    const int m0 = blockIdx.y * GBM, n0 = blockIdx.x * GBN;
    const int rsel = lane & 15;
    const bool hi = lane >= 16;

    // per-thread staging assignment: 16B chunk (row r, cols col..col+7)
    const int r   = tid >> 2;          // 0..63
    const int col = (tid & 3) * 8;     // 0,8,16,24
    const u16* ga0 = A  + (size_t)(m0 + r) * K + col;        // A rows 0..63
    const u16* ga1 = A  + (size_t)(m0 + r + 64) * K + col;   // A rows 64..127
    const u16* gb  = Bt + (size_t)(n0 + r) * K + col;        // Bt rows 0..63
    unsigned la0[2], la1[2], lb[2];
    #pragma unroll
    for (int bi = 0; bi < 2; bi++) {
        la0[bi] = (unsigned)(size_t)&As[bi][r][col];
        la1[bi] = (unsigned)(size_t)&As[bi][r + 64][col];
        lb[bi]  = (unsigned)(size_t)&Bs[bi][r][col];
    }

    v8f acc[2][2];
    #pragma unroll
    for (int i = 0; i < 2; i++)
        #pragma unroll
        for (int j = 0; j < 2; j++) acc[i][j] = zero8();

    const int nk = K >> 5;             // K / 32

    // prime buffer 0
    async_b128(la0[0], ga0);
    async_b128(la1[0], ga1);
    async_b128(lb[0],  gb);
    wait_async0();
    __syncthreads();

    int buf = 0;
    for (int kk = 0; kk < nk; ++kk) {
        // prefetch next tile into the other buffer (overlaps the WMMAs below)
        if (kk + 1 < nk) {
            size_t koff = (size_t)(kk + 1) * GBK;
            async_b128(la0[buf ^ 1], ga0 + koff);
            async_b128(la1[buf ^ 1], ga1 + koff);
            async_b128(lb[buf ^ 1],  gb + koff);
        }

        Frag af[2], bfr[2];
        const int abase = hi ? 8 : 0;    // A frag: K {0-7,16-23} / {8-15,24-31}
        const int bbase = hi ? 8 : 0;    // B frag: K {0-7,16-23} / {8-15,24-31}
        #pragma unroll
        for (int i = 0; i < 2; i++) {
            int row = wm * 32 + i * 16 + rsel;
            af[i].u[0] = *(const u32x4*)&As[buf][row][abase];
            af[i].u[1] = *(const u32x4*)&As[buf][row][abase + 16];
        }
        #pragma unroll
        for (int j = 0; j < 2; j++) {
            int row = wn * 32 + j * 16 + rsel;
            bfr[j].u[0] = *(const u32x4*)&Bs[buf][row][bbase];
            bfr[j].u[1] = *(const u32x4*)&Bs[buf][row][bbase + 16];
        }
        #pragma unroll
        for (int i = 0; i < 2; i++)
            #pragma unroll
            for (int j = 0; j < 2; j++)
                acc[i][j] = __builtin_amdgcn_wmma_f32_16x16x32_bf16(
                    false, af[i].v, false, bfr[j].v, (short)0, acc[i][j],
                    false, false);

        wait_async0();      // next buffer's DMA complete
        __syncthreads();    // all waves done reading current buffer
        buf ^= 1;
    }

    // epilogue: C rows = r (lanes 0-15) / r+8 (lanes 16-31), col = lane&15
    const int colbase = n0 + wn * 32 + rsel;
    const int rowadd = hi ? 8 : 0;
    #pragma unroll
    for (int i = 0; i < 2; i++) {
        int rbase = m0 + wm * 32 + i * 16 + rowadd;
        #pragma unroll
        for (int j = 0; j < 2; j++) {
            int c2 = colbase + j * 16;
            #pragma unroll
            for (int rr = 0; rr < 8; rr++)
                C[(size_t)(rbase + rr) * N + c2] = acc[i][j][rr];
        }
    }
}

// ---------------------------------------------------------------------------
// RMSNorm over fp32 rows (stride in_stride), write bf16 (stride dim)
// ---------------------------------------------------------------------------
__global__ __launch_bounds__(256) void rmsnorm_to_bf16(
        const float* __restrict__ x, const float* __restrict__ w,
        u16* __restrict__ y, int in_stride, int dim) {
    const int row = blockIdx.x;
    const float* xr = x + (size_t)row * in_stride;
    float ss = 0.f;
    for (int i = threadIdx.x; i < dim; i += 256) { float v = xr[i]; ss += v * v; }
    #pragma unroll
    for (int off = 16; off >= 1; off >>= 1) ss += __shfl_xor(ss, off, 32);
    __shared__ float red[8];
    int lane = threadIdx.x & 31, wid = threadIdx.x >> 5;
    if (lane == 0) red[wid] = ss;
    __syncthreads();
    if (threadIdx.x == 0) {
        float t = 0.f;
        #pragma unroll
        for (int i = 0; i < 8; i++) t += red[i];
        red[0] = rsqrtf(t / (float)dim + 1e-6f);
    }
    __syncthreads();
    float rinv = red[0];
    u16* yr = y + (size_t)row * dim;
    for (int i = threadIdx.x; i < dim; i += 256)
        yr[i] = f2bf(xr[i] * rinv * w[i]);
}

// ---------------------------------------------------------------------------
// Interleaved RoPE (matches reference deinterleave + half-rotation):
//   out[j]    = x[2j]*cos - x[2j+1]*sin        (j < 32)
//   out[j+32] = x[2j+1]*cos + x[2j]*sin
// inv_freq[j] = theta^(-2j/64) = exp(-j * ln(theta)/32)
// ---------------------------------------------------------------------------
__device__ inline void rope_pair(int i, int t, float even, float odd,
                                 float* e_lo, float* e_hi) {
    float inv = __expf(-(float)i * 0.4100738544464111f); // ln(500000)/32
    float ang = (float)t * inv;
    float c = __cosf(ang), s = __sinf(ang);
    *e_lo = even * c - odd * s;
    *e_hi = odd * c + even * s;
}

// q fp32 [B,T,H,192] -> Qf bf16 [B*H][T][192] (RoPE applied to last 64)
__global__ void build_q_kernel(const float* __restrict__ q,
                               u16* __restrict__ Qf, int Tn) {
    int idx = blockIdx.x;
    int h = idx & 15;
    int t = (idx >> 4) % Tn;
    int b = idx / (16 * Tn);
    const float* src = q + (((size_t)(b * Tn + t)) * 16 + h) * 192;
    u16* dst = Qf + (((size_t)(b * 16 + h)) * Tn + t) * 192;
    int d = threadIdx.x;                 // 64 threads
    dst[d]      = f2bf(src[d]);
    dst[d + 64] = f2bf(src[d + 64]);
    if (d < 32) {
        float lo, hi2;
        rope_pair(d, t, src[128 + 2 * d], src[128 + 2 * d + 1], &lo, &hi2);
        dst[128 + d] = f2bf(lo);
        dst[160 + d] = f2bf(hi2);
    }
}

// ckv fp32 [B,T,576] (rope part), kv fp32 [B,T,H,256] ->
//   Kf bf16 [B*H][T][192]  (k_nope ++ roped k_rope broadcast over heads)
//   Vt bf16 [B*H][128][T]  (transposed V for WMMA B-fragments)
__global__ __launch_bounds__(256) void build_kv_kernel(
        const float* __restrict__ ckv, const float* __restrict__ kv,
        u16* __restrict__ Kfo, u16* __restrict__ Vto, int Tn) {
    int t = blockIdx.x % Tn;
    int b = blockIdx.x / Tn;
    __shared__ float kr[64];
    if (threadIdx.x < 32) {
        int i = threadIdx.x;
        const float* rsrc = ckv + (size_t)(b * Tn + t) * 576 + 512;
        float lo, hi2;
        rope_pair(i, t, rsrc[2 * i], rsrc[2 * i + 1], &lo, &hi2);
        kr[i] = lo;
        kr[32 + i] = hi2;
    }
    __syncthreads();
    const float* kvrow = kv + (size_t)(b * Tn + t) * 4096;
    for (int i = threadIdx.x; i < 16 * 192; i += 256) {
        int h = i / 192, d = i % 192;
        float v = (d < 128) ? kvrow[h * 256 + d] : kr[d - 128];
        Kfo[(((size_t)(b * 16 + h)) * Tn + t) * 192 + d] = f2bf(v);
    }
    for (int i = threadIdx.x; i < 16 * 128; i += 256) {
        int h = i / 128, d = i % 128;
        Vto[(((size_t)(b * 16 + h)) * 128 + d) * Tn + t] =
            f2bf(kvrow[h * 256 + 128 + d]);
    }
}

// ---------------------------------------------------------------------------
// Flash attention, causal, d_qk=192, d_v=128.
// Grid: (T/128, B*H). 8 waves/block, each wave owns 16 query rows.
// Per 32-key chunk: 12 WMMA for S = Q K^T, online softmax, P staged through
// LDS into A-fragment shape, 8 WMMA for O += P V.
// ---------------------------------------------------------------------------
__global__ __launch_bounds__(256) void mla_attn_kernel(
        const u16* __restrict__ Qf, const u16* __restrict__ Kf,
        const u16* __restrict__ Vt, u16* __restrict__ Out, int Tn) {
    __shared__ u16 Pld[8][16][40];       // per-wave 16x32 P patch (padded)

    const int lane = threadIdx.x & 31, wid = threadIdx.x >> 5;
    const int bh = blockIdx.y;
    const int b = bh >> 4, h = bh & 15;
    const int q0 = blockIdx.x * 128 + wid * 16;
    const int rsel = lane & 15;
    const bool hi = lane >= 16;

    const u16* Qb = Qf + (size_t)bh * Tn * 192;
    const u16* Kb = Kf + (size_t)bh * Tn * 192;
    const u16* Vb = Vt + (size_t)bh * 128 * Tn;

    // Q A-fragments for all 6 K-dim chunks (d = 0..191)
    Frag qfr[6];
    {
        const u16* qrow = Qb + (size_t)(q0 + rsel) * 192;
        const int abase = hi ? 8 : 0;
        #pragma unroll
        for (int c = 0; c < 6; c++) {
            qfr[c].u[0] = *(const u32x4*)(qrow + c * 32 + abase);
            qfr[c].u[1] = *(const u32x4*)(qrow + c * 32 + abase + 16);
        }
    }

    v8f o[8];
    #pragma unroll
    for (int i = 0; i < 8; i++) o[i] = zero8();
    float mrow[8], lrow[8];
    #pragma unroll
    for (int r = 0; r < 8; r++) { mrow[r] = -1e30f; lrow[r] = 0.f; }

    const float sc = 0.07216878364870323f;   // 192^-0.5
    const int kend = q0 + 16;                // keys 0 .. q0+15

    for (int k0 = 0; k0 < kend; k0 += 32) {
        // ---- scores: two 16x16 tiles over 6 d-chunks ----
        v8f s0 = zero8(), s1 = zero8();
        const int bbase = hi ? 8 : 0;
        #pragma unroll
        for (int c = 0; c < 6; c++) {
            Frag kf0, kf1;
            const u16* kr0 = Kb + (size_t)(k0 + rsel) * 192 + c * 32 + bbase;
            const u16* kr1 = Kb + (size_t)(k0 + 16 + rsel) * 192 + c * 32 + bbase;
            kf0.u[0] = *(const u32x4*)kr0;
            kf0.u[1] = *(const u32x4*)(kr0 + 16);
            kf1.u[0] = *(const u32x4*)kr1;
            kf1.u[1] = *(const u32x4*)(kr1 + 16);
            s0 = __builtin_amdgcn_wmma_f32_16x16x32_bf16(
                false, qfr[c].v, false, kf0.v, (short)0, s0, false, false);
            s1 = __builtin_amdgcn_wmma_f32_16x16x32_bf16(
                false, qfr[c].v, false, kf1.v, (short)0, s1, false, false);
        }

        // ---- scale + causal mask ----
        const int colg0 = k0 + rsel, colg1 = colg0 + 16;
        const int rowg = q0 + (hi ? 8 : 0);
        const bool needmask = (k0 + 31 > q0);
        #pragma unroll
        for (int r = 0; r < 8; r++) {
            float a = s0[r] * sc, bv = s1[r] * sc;
            if (needmask) {
                if (colg0 > rowg + r) a = -1e30f;
                if (colg1 > rowg + r) bv = -1e30f;
            }
            s0[r] = a; s1[r] = bv;
        }

        // ---- online softmax (rows striped across 16-lane halves) ----
        float corr[8];
        #pragma unroll
        for (int r = 0; r < 8; r++) {
            float mx = fmaxf(s0[r], s1[r]);
            #pragma unroll
            for (int off = 8; off >= 1; off >>= 1)
                mx = fmaxf(mx, __shfl_xor(mx, off, 32));
            float mnew = fmaxf(mrow[r], mx);
            corr[r] = __expf(mrow[r] - mnew);
            mrow[r] = mnew;
            float p0 = __expf(s0[r] - mnew);
            float p1 = __expf(s1[r] - mnew);
            s0[r] = p0; s1[r] = p1;
            float sum = p0 + p1;
            #pragma unroll
            for (int off = 8; off >= 1; off >>= 1)
                sum += __shfl_xor(sum, off, 32);
            lrow[r] = lrow[r] * corr[r] + sum;
        }
        #pragma unroll
        for (int nt = 0; nt < 8; nt++)
            #pragma unroll
            for (int r = 0; r < 8; r++) o[nt][r] *= corr[r];

        // ---- stage P to LDS, reload in A-fragment shape ----
        #pragma unroll
        for (int r = 0; r < 8; r++) {
            int row = r + (hi ? 8 : 0);
            Pld[wid][row][rsel]      = f2bf(s0[r]);
            Pld[wid][row][16 + rsel] = f2bf(s1[r]);
        }
        Frag pf;
        {
            const int abase = hi ? 8 : 0;
            pf.u[0] = *(const u32x4*)&Pld[wid][rsel][abase];
            pf.u[1] = *(const u32x4*)&Pld[wid][rsel][abase + 16];
        }

        // ---- O += P @ V over 8 d-tiles ----
        const int vkb = k0 + (hi ? 8 : 0);
        #pragma unroll
        for (int nt = 0; nt < 8; nt++) {
            Frag vf;
            const u16* vr = Vb + (size_t)(nt * 16 + rsel) * Tn + vkb;
            vf.u[0] = *(const u32x4*)vr;
            vf.u[1] = *(const u32x4*)(vr + 16);
            o[nt] = __builtin_amdgcn_wmma_f32_16x16x32_bf16(
                false, pf.v, false, vf.v, (short)0, o[nt], false, false);
        }
    }

    // ---- normalize + write bf16 [B][T][H*128] ----
    const int rowadd = hi ? 8 : 0;
    #pragma unroll
    for (int nt = 0; nt < 8; nt++) {
        int d = nt * 16 + rsel;
        #pragma unroll
        for (int r = 0; r < 8; r++) {
            int row = q0 + r + rowadd;
            Out[((size_t)(b * Tn) + row) * 2048 + h * 128 + d] =
                f2bf(o[nt][r] / lrow[r]);
        }
    }
}

// ---------------------------------------------------------------------------
// Host launcher
// ---------------------------------------------------------------------------
extern "C" void kernel_launch(void* const* d_in, const int* in_sizes, int n_in,
                              void* d_out, int out_size, void* d_ws,
                              size_t ws_size, hipStream_t stream) {
    (void)in_sizes; (void)n_in; (void)out_size; (void)ws_size;

    const float* x       = (const float*)d_in[0];
    const float* q_a_w   = (const float*)d_in[1];
    const float* q_a_ln  = (const float*)d_in[2];
    const float* q_b_w   = (const float*)d_in[3];
    const float* kv_a_w  = (const float*)d_in[4];
    const float* kv_a_ln = (const float*)d_in[5];
    const float* kv_b_w  = (const float*)d_in[6];
    const float* o_w     = (const float*)d_in[7];
    float* out = (float*)d_out;

    const int Bn = 2, Tn = 2048, M = Bn * Tn;   // M = 4096

    char* base = (char*)d_ws;
    size_t off = 0;
    auto alloc = [&](size_t bytes) -> void* {
        void* p = base + off;
        off += (bytes + 255) & ~(size_t)255;
        return p;
    };

    u16*   xb    = (u16*)  alloc((size_t)M * 2048 * 2);
    u16*   wqat  = (u16*)  alloc((size_t)1536 * 2048 * 2);
    u16*   wqbt  = (u16*)  alloc((size_t)3072 * 1536 * 2);
    u16*   wkvat = (u16*)  alloc((size_t)576 * 2048 * 2);
    u16*   wkvbt = (u16*)  alloc((size_t)4096 * 512 * 2);
    u16*   wot   = (u16*)  alloc((size_t)2048 * 2048 * 2);
    float* qa    = (float*)alloc((size_t)M * 1536 * 4);
    u16*   qan   = (u16*)  alloc((size_t)M * 1536 * 2);
    float* qq    = (float*)alloc((size_t)M * 3072 * 4);
    float* ckv   = (float*)alloc((size_t)M * 576 * 4);
    u16*   kvcn  = (u16*)  alloc((size_t)M * 512 * 2);
    float* kvf   = (float*)alloc((size_t)M * 4096 * 4);
    u16*   Qfb   = (u16*)  alloc((size_t)32 * Tn * 192 * 2);
    u16*   Kfb   = (u16*)  alloc((size_t)32 * Tn * 192 * 2);
    u16*   Vtb   = (u16*)  alloc((size_t)32 * 128 * Tn * 2);
    u16*   attn  = (u16*)  alloc((size_t)M * 2048 * 2);

    dim3 blk(256);

    // 1) bf16 conversions
    {
        size_t n = (size_t)M * 2048;
        f32_to_bf16_kernel<<<(unsigned)((n + 255) / 256), blk, 0, stream>>>(x, xb, n);
    }
    transpose_f32_to_bf16<<<dim3(1536 / 32, 2048 / 32), blk, 0, stream>>>(q_a_w,  wqat,  2048, 1536);
    transpose_f32_to_bf16<<<dim3(3072 / 32, 1536 / 32), blk, 0, stream>>>(q_b_w,  wqbt,  1536, 3072);
    transpose_f32_to_bf16<<<dim3(576  / 32, 2048 / 32), blk, 0, stream>>>(kv_a_w, wkvat, 2048, 576);
    transpose_f32_to_bf16<<<dim3(4096 / 32, 512  / 32), blk, 0, stream>>>(kv_b_w, wkvbt, 512,  4096);
    transpose_f32_to_bf16<<<dim3(2048 / 32, 2048 / 32), blk, 0, stream>>>(o_w,    wot,   2048, 2048);

    // 2) q path: q_a = x@W_qa -> rmsnorm -> q = .@W_qb
    gemm_bf16_f32<<<dim3(1536 / GBN, M / GBM), blk, 0, stream>>>(xb, wqat, qa, M, 1536, 2048);
    rmsnorm_to_bf16<<<M, blk, 0, stream>>>(qa, q_a_ln, qan, 1536, 1536);
    gemm_bf16_f32<<<dim3(3072 / GBN, M / GBM), blk, 0, stream>>>(qan, wqbt, qq, M, 3072, 1536);

    // 3) kv path: ckv = x@W_kva -> rmsnorm(512) -> kv = .@W_kvb
    gemm_bf16_f32<<<dim3(576 / GBN, M / GBM), blk, 0, stream>>>(xb, wkvat, ckv, M, 576, 2048);
    rmsnorm_to_bf16<<<M, blk, 0, stream>>>(ckv, kv_a_ln, kvcn, 576, 512);
    gemm_bf16_f32<<<dim3(4096 / GBN, M / GBM), blk, 0, stream>>>(kvcn, wkvbt, kvf, M, 4096, 512);

    // 4) RoPE + head-major staging
    build_q_kernel<<<Bn * Tn * 16, 64, 0, stream>>>(qq, Qfb, Tn);
    build_kv_kernel<<<Bn * Tn, blk, 0, stream>>>(ckv, kvf, Kfb, Vtb, Tn);

    // 5) causal flash attention
    mla_attn_kernel<<<dim3(Tn / 128, Bn * 16), blk, 0, stream>>>(Qfb, Kfb, Vtb, attn, Tn);

    // 6) output projection -> fp32 d_out
    gemm_bf16_f32<<<dim3(2048 / GBN, M / GBM), blk, 0, stream>>>(attn, wot, out, M, 2048, 2048);
}
